// GroupedGemmRunner_20444044329133
// MI455X (gfx1250) — compile-verified, matrix operand order checked by reading, not run on hardware
//
#include <hip/hip_runtime.h>

// ---------------- problem constants (from reference) ----------------
#define E_EXPERTS 8
#define KDIM 2048
#define NDIM 4096
// ---------------- tiling ----------------
#define BM 128               // M rows per block tile
#define BN 256               // N cols per block tile
#define BK 32                // bf16 WMMA K depth per step
#define BKP 40               // padded row length in ushorts (80B: 16B-aligned, bank-conflict-free)
#define NTILES (NDIM / BN)   // 16 n-tiles

typedef __attribute__((ext_vector_type(16))) __bf16 v16bf;
typedef __attribute__((ext_vector_type(8)))  float  v8f;
typedef unsigned int  u32x4 __attribute__((ext_vector_type(4)));
typedef unsigned int  u32x2 __attribute__((ext_vector_type(2)));
typedef float         f32x4 __attribute__((ext_vector_type(4)));

union Frag {
    v16bf v;
    u32x4 q[2];
};

// split float4 into bf16 hi/lo (native v_cvt, RNE) and store packed pairs to LDS
__device__ __forceinline__ void cvt_store(unsigned short* hiP, unsigned short* loP, f32x4 v) {
    union P2 { __bf16 h[2]; unsigned u; };
    __bf16 h0 = (__bf16)v[0], h1 = (__bf16)v[1], h2 = (__bf16)v[2], h3 = (__bf16)v[3];
    __bf16 l0 = (__bf16)(v[0] - (float)h0);
    __bf16 l1 = (__bf16)(v[1] - (float)h1);
    __bf16 l2 = (__bf16)(v[2] - (float)h2);
    __bf16 l3 = (__bf16)(v[3] - (float)h3);
    P2 a, b, c, d;
    a.h[0] = h0; a.h[1] = h1;  b.h[0] = h2; b.h[1] = h3;
    c.h[0] = l0; c.h[1] = l1;  d.h[0] = l2; d.h[1] = l3;
    u32x2 hp, lp;
    hp[0] = a.u; hp[1] = b.u;
    lp[0] = c.u; lp[1] = d.u;
    *(u32x2*)hiP = hp;
    *(u32x2*)loP = lp;
}

__global__ __launch_bounds__(256)
void grouped_gemm_bf16x2_wmma(const float* __restrict__ A,
                              const float* __restrict__ B,
                              float* __restrict__ C,
                              const long long* __restrict__ seg_indptr,
                              const long long* __restrict__ widx,
                              int T) {
    // double-buffered hi/lo planes: A = 40KB, B = 80KB (of 320KB WGP LDS)
    __shared__ unsigned short sA[2][2][BM][BKP];   // [buf][hi/lo][row(m)][k]
    __shared__ unsigned short sB[2][2][BN][BKP];   // [buf][hi/lo][row(n)][k]

    // ---- locate (segment, m-tile, n-tile) for this block ----
    int tile = blockIdx.x;
    long long segs = 0, sege = 0;
    int s = 0;
    bool found = false;
    for (s = 0; s < E_EXPERTS; ++s) {
        segs = seg_indptr[s];
        sege = seg_indptr[s + 1];
        int len = (int)(sege - segs);
        int mt = (len + BM - 1) / BM;
        int cnt = mt * NTILES;
        if (tile < cnt) { found = true; break; }
        tile -= cnt;
    }
    if (!found) return;   // block-uniform exit

    const int mti = tile / NTILES;
    const int nti = tile % NTILES;
    const long long m0 = segs + (long long)mti * BM;
    const long long remRows = sege - m0;
    const int rows = remRows < (long long)BM ? (int)remRows : BM;
    const int gn = nti * BN;
    const float* __restrict__ Bexp = B + (size_t)widx[s] * (size_t)NDIM * (size_t)KDIM;

    const int tid  = threadIdx.x;
    const int lane = tid & 31;
    const int wave = tid >> 5;
    const int wm   = wave >> 2;     // 0..1 : 64-row slice of the 128-row tile
    const int wn   = wave & 3;      // 0..3 : 64-col slice of the 256-col tile
    const int half = lane >> 4;     // lane group for WMMA k-split
    const int l16  = lane & 15;

    // per-lane LDS fragment offsets (ushort units) within one buffer
    int abase[4], bbase[4];
#pragma unroll
    for (int mi = 0; mi < 4; ++mi)
        abase[mi] = (wm * 64 + mi * 16 + l16) * BKP + half * 8;
#pragma unroll
    for (int ni = 0; ni < 4; ++ni)
        bbase[ni] = (wn * 64 + ni * 16 + l16) * BKP + half * 8;

    // global load mapping (unconditional, row-clamped; invalid rows zeroed by cndmask)
    //   A: 4 x f32x4 per thread over 128 rows x 8 quads
    //   B: 8 x f32x4 per thread over 256 rows x 8 quads
    int  arow[4], acol[4];  bool avalid[4];
    int  brow[8], bcol[8];
#pragma unroll
    for (int i = 0; i < 4; ++i) {
        int idx = tid + i * 256;
        int r = idx >> 3;
        arow[i] = r < rows ? r : 0;      // clamp: stay in-bounds
        avalid[i] = r < rows;
        acol[i] = (idx & 7) * 4;
    }
#pragma unroll
    for (int i = 0; i < 8; ++i) {
        int idx = tid + i * 256;
        brow[i] = idx >> 3;
        bcol[i] = (idx & 7) * 4;
    }

    v8f acc[4][4];
#pragma unroll
    for (int mi = 0; mi < 4; ++mi)
#pragma unroll
        for (int ni = 0; ni < 4; ++ni)
#pragma unroll
            for (int i = 0; i < 8; ++i) acc[mi][ni][i] = 0.0f;

    // ---- prologue: stage k-step 0 into registers ----
    f32x4 av[4], bv[8];
#pragma unroll
    for (int i = 0; i < 4; ++i) {
        f32x4 t = *(const f32x4*)&A[(m0 + arow[i]) * (long long)KDIM + acol[i]];
        f32x4 z; z[0] = z[1] = z[2] = z[3] = 0.0f;
        av[i] = avalid[i] ? t : z;
    }
#pragma unroll
    for (int i = 0; i < 8; ++i)
        bv[i] = *(const f32x4*)&Bexp[(size_t)(gn + brow[i]) * KDIM + bcol[i]];

    const int nk = KDIM / BK;   // 64 k-steps
    int p = 0;
    for (int kt = 0; kt < nk; ++kt) {
        // ---- convert + stage current k-step into LDS buffer p ----
#pragma unroll
        for (int i = 0; i < 4; ++i)
            cvt_store(&sA[p][0][arow[i]][acol[i]], &sA[p][1][arow[i]][acol[i]], av[i]);
#pragma unroll
        for (int i = 0; i < 8; ++i)
            cvt_store(&sB[p][0][brow[i]][bcol[i]], &sB[p][1][brow[i]][bcol[i]], bv[i]);

        // ---- issue global loads for next k-step (overlaps compute below) ----
        if (kt + 1 < nk) {
            const int kbase = (kt + 1) * BK;
#pragma unroll
            for (int i = 0; i < 4; ++i) {
                f32x4 t = *(const f32x4*)&A[(m0 + arow[i]) * (long long)KDIM + kbase + acol[i]];
                f32x4 z; z[0] = z[1] = z[2] = z[3] = 0.0f;
                av[i] = avalid[i] ? t : z;
            }
#pragma unroll
            for (int i = 0; i < 8; ++i)
                bv[i] = *(const f32x4*)&Bexp[(size_t)(gn + brow[i]) * KDIM + kbase + bcol[i]];
        }

        __syncthreads();   // stores to buf p visible; prev iter's reads of buf p^1 done

        const unsigned short* A0 = &sA[p][0][0][0];
        const unsigned short* A1 = &sA[p][1][0][0];
        const unsigned short* B0 = &sB[p][0][0][0];
        const unsigned short* B1 = &sB[p][1][0][0];

        // ---- A fragments: 4 m-frags, hi+lo ----
        Frag ah[4], al[4];
#pragma unroll
        for (int mi = 0; mi < 4; ++mi) {
            ah[mi].q[0] = *(const u32x4*)(A0 + abase[mi]);
            ah[mi].q[1] = *(const u32x4*)(A0 + abase[mi] + 16);
            al[mi].q[0] = *(const u32x4*)(A1 + abase[mi]);
            al[mi].q[1] = *(const u32x4*)(A1 + abase[mi] + 16);
        }

        // ---- per n-frag: load B hi/lo, then 12 WMMAs (hi*hi + hi*lo + lo*hi) ----
#pragma unroll
        for (int ni = 0; ni < 4; ++ni) {
            Frag bh, bl;
            bh.q[0] = *(const u32x4*)(B0 + bbase[ni]);
            bh.q[1] = *(const u32x4*)(B0 + bbase[ni] + 16);
            bl.q[0] = *(const u32x4*)(B1 + bbase[ni]);
            bl.q[1] = *(const u32x4*)(B1 + bbase[ni] + 16);
#pragma unroll
            for (int mi = 0; mi < 4; ++mi) {
                acc[mi][ni] = __builtin_amdgcn_wmma_f32_16x16x32_bf16(
                    false, ah[mi].v, false, bh.v, (short)0, acc[mi][ni], false, false);
                acc[mi][ni] = __builtin_amdgcn_wmma_f32_16x16x32_bf16(
                    false, ah[mi].v, false, bl.v, (short)0, acc[mi][ni], false, false);
                acc[mi][ni] = __builtin_amdgcn_wmma_f32_16x16x32_bf16(
                    false, al[mi].v, false, bh.v, (short)0, acc[mi][ni], false, false);
            }
        }
        p ^= 1;
    }

    // ---- epilogue: 16x16 f32 C/D layout -> global ----
#pragma unroll
    for (int mi = 0; mi < 4; ++mi) {
#pragma unroll
        for (int ni = 0; ni < 4; ++ni) {
            const int mrow0 = wm * 64 + mi * 16 + 8 * half;
            const int ncol  = gn + wn * 64 + ni * 16 + l16;
#pragma unroll
            for (int i = 0; i < 8; ++i) {
                const long long grow = m0 + mrow0 + i;
                if (grow < sege)
                    C[grow * (long long)NDIM + ncol] = acc[mi][ni][i];
            }
        }
    }
}

extern "C" void kernel_launch(void* const* d_in, const int* in_sizes, int n_in,
                              void* d_out, int out_size, void* d_ws, size_t ws_size,
                              hipStream_t stream) {
    // setup_inputs order: a, b, c, batch_size, weight_column_major, seg_indptr, weight_indices
    const float*     a    = (const float*)d_in[0];
    const float*     b    = (const float*)d_in[1];
    const long long* seg  = (const long long*)d_in[5];   // int64
    const long long* widx = (const long long*)d_in[6];   // int64
    float* c = (float*)d_out;

    const int T = in_sizes[0] / KDIM;
    // upper bound on tiles: per-segment rounding adds at most E extra m-tiles
    const int maxMTiles = (T + BM - 1) / BM + E_EXPERTS;
    const int grid = maxMTiles * NTILES;

    grouped_gemm_bf16x2_wmma<<<grid, 256, 0, stream>>>(a, b, c, seg, widx, T);
}
